// StructureLayer_18760417149442
// MI455X (gfx1250) — matrix operand order repaired and checked by reading, not compiled, not run on hardware
//
#include <hip/hip_runtime.h>
#include <math.h>

// ---------------------------------------------------------------------------
// StructureLayer (AlphaFold-style IPA + transition) for gfx1250 / MI455X.
// All large contractions run through v_wmma_f32_16x16x32_bf16 (fp32 accum).
// Each wave computes a 16x32 output tile (A fragment reused across two WMMAs).
// Inner GEMM loops are branch-free: OOB rows/cols are clamped (their garbage
// only reaches outputs that the store guards discard), K is always a multiple
// of 32 (point-dot operands are zero-padded to K=32).
// ---------------------------------------------------------------------------

#define NRES 768
#define CS   256
#define CZ   128
#define CH   128
#define NH   8
#define PQK  8
#define PV   12
#define CONCAT_DIM 2432   // H*CH + 3*H*PV + H*PV + H*CZ = 1024+288+96+1024

typedef __attribute__((ext_vector_type(16))) __bf16 v16bf;
typedef __attribute__((ext_vector_type(8)))  float  v8f;

__device__ __forceinline__ __bf16 f2bf(float f) {
  return (__bf16)f;   // native RNE convert (hardware bf16 cvt on gfx1250)
}

__device__ __forceinline__ void put4(v16bf& v, int o, float4 f) {
  v[o + 0] = f2bf(f.x); v[o + 1] = f2bf(f.y);
  v[o + 2] = f2bf(f.z); v[o + 3] = f2bf(f.w);
}

// Build an A fragment (contiguous k order: half h holds k = 16h + s) from a K-contiguous row.
__device__ __forceinline__ v16bf load_frag_rowmajor(const float* row, int k0) {
  const float4* p4 = (const float4*)(row + k0);
  float4 F0 = p4[0], F1 = p4[1], F2 = p4[2], F3 = p4[3]; // K: +0..15
  v16bf v;
  put4(v, 0, F0); put4(v, 4, F1); put4(v, 8, F2); put4(v, 12, F3);
  return v;
}

// Build a B fragment (contiguous k order: half h holds k = 16h + s) from a K-contiguous row (W^T case).
__device__ __forceinline__ v16bf load_fragb_rowmajor(const float* row, int k0) {
  const float4* p4 = (const float4*)(row + k0);
  float4 F0 = p4[0], F1 = p4[1], F2 = p4[2], F3 = p4[3]; // K: +0..15
  v16bf v;
  put4(v, 0, F0); put4(v, 4, F1); put4(v, 8, F2); put4(v, 12, F3);
  return v;
}

// ---------------------------------------------------------------------------
// GEMM 1: out[m,n] = scale * sum_k A[m,k] * W[n,k]  (+bias[n]) (+relu)
// Both operands K-contiguous. REQUIRES: K % 32 == 0; all row strides and
// batch offsets multiples of 4 floats (16B-aligned float4 loads).
// One 16x32 output tile per wave; batched via blockIdx.z.
// ---------------------------------------------------------------------------
__global__ __launch_bounds__(256) void gemm_wt(
    const float* __restrict__ A, long lda, long bA,
    const float* __restrict__ W, long ldw, long bW,
    const float* __restrict__ bias,
    float* __restrict__ O, long ldo, long bO,
    int M, int Nt, int K, float scale, int relu)
{
  int lane = threadIdx.x & 31;
  int ntl = (Nt + 15) >> 4;
  int ntp = (ntl + 1) >> 1;                  // pairs of 16-col sub-tiles
  long tiles = (long)((M + 15) >> 4) * ntp;
  long tile  = (long)blockIdx.x * 8 + (threadIdx.x >> 5);
  if (tile >= tiles) return;                 // wave-uniform exit (EXEC all-1s)
  int tm = (int)(tile / ntp), tp = (int)(tile % ntp);
  int row0 = tm << 4, col0 = tp << 5;
  bool two = (tp * 2 + 1) < ntl;             // wave-uniform
  int z = blockIdx.z;

  int lr = lane & 15;                // A-row / W-row / D-col within tile
  int ka = (lane & 16) ? 16 : 0;     // A K-offset split: half h holds k = 16h + s
  int kb = (lane & 16) ? 16 : 0;     // B K-offset split: half h holds k = 16h + s
  int arow  = row0 + lr;      if (arow  > M - 1)  arow  = M - 1;  // clamp
  int wrow0 = col0 + lr;      if (wrow0 > Nt - 1) wrow0 = Nt - 1;
  int wrow1 = col0 + 16 + lr; if (wrow1 > Nt - 1) wrow1 = Nt - 1;
  const float* Arow = A + (long)z * bA + (long)arow  * lda;
  const float* Wr0  = W + (long)z * bW + (long)wrow0 * ldw;
  const float* Wr1  = W + (long)z * bW + (long)wrow1 * ldw;
  float*       Ob   = O + (long)z * bO;

  v8f c0, c1;
#pragma unroll
  for (int r = 0; r < 8; ++r) { c0[r] = 0.0f; c1[r] = 0.0f; }

  if (two) {
    for (int k0 = 0; k0 < K; k0 += 32) {
      v16bf a  = load_frag_rowmajor(Arow, k0 + ka);
      v16bf b0 = load_fragb_rowmajor(Wr0, k0 + kb);
      v16bf b1 = load_fragb_rowmajor(Wr1, k0 + kb);
      c0 = __builtin_amdgcn_wmma_f32_16x16x32_bf16(false, a, false, b0,
                                                   (short)0, c0, false, false);
      c1 = __builtin_amdgcn_wmma_f32_16x16x32_bf16(false, a, false, b1,
                                                   (short)0, c1, false, false);
    }
  } else {
    for (int k0 = 0; k0 < K; k0 += 32) {
      v16bf a  = load_frag_rowmajor(Arow, k0 + ka);
      v16bf b0 = load_fragb_rowmajor(Wr0, k0 + kb);
      c0 = __builtin_amdgcn_wmma_f32_16x16x32_bf16(false, a, false, b0,
                                                   (short)0, c0, false, false);
    }
  }

  int rb = (lane & 16) ? 8 : 0;
  int colA = col0 + lr;
  if (colA < Nt) {
    float bv = bias ? bias[colA] : 0.0f;
#pragma unroll
    for (int r = 0; r < 8; ++r) {
      int row = row0 + rb + r;
      if (row < M) {
        float v = c0[r] * scale + bv;
        if (relu) v = fmaxf(v, 0.0f);
        Ob[(long)row * ldo + colA] = v;
      }
    }
  }
  int colB = col0 + 16 + lr;
  if (two && colB < Nt) {
    float bv = bias ? bias[colB] : 0.0f;
#pragma unroll
    for (int r = 0; r < 8; ++r) {
      int row = row0 + rb + r;
      if (row < M) {
        float v = c1[r] * scale + bv;
        if (relu) v = fmaxf(v, 0.0f);
        Ob[(long)row * ldo + colB] = v;
      }
    }
  }
}

// ---------------------------------------------------------------------------
// GEMM 2: out[m,n] = sum_k A[m,k] * B[k,n]   (B row-major K x N, stride ldb)
// REQUIRES: K % 32 == 0. B columns clamped; loads lane-coalesced per k-row.
// One 16x32 output tile per wave; batched via blockIdx.z.
// ---------------------------------------------------------------------------
__global__ __launch_bounds__(256) void gemm_bn(
    const float* __restrict__ A, long lda, long bA,
    const float* __restrict__ Bm, long ldb, long bB,
    float* __restrict__ O, long ldo, long bO,
    int M, int Nt, int K)
{
  int lane = threadIdx.x & 31;
  int ntl = (Nt + 15) >> 4;
  int ntp = (ntl + 1) >> 1;
  long tiles = (long)((M + 15) >> 4) * ntp;
  long tile  = (long)blockIdx.x * 8 + (threadIdx.x >> 5);
  if (tile >= tiles) return;
  int tm = (int)(tile / ntp), tp = (int)(tile % ntp);
  int row0 = tm << 4, col0 = tp << 5;
  bool two = (tp * 2 + 1) < ntl;
  int z = blockIdx.z;

  int lr = lane & 15;
  int ka = (lane & 16) ? 16 : 0;
  int kb = (lane & 16) ? 16 : 0;
  int arow  = row0 + lr;      if (arow  > M - 1)  arow  = M - 1;
  int bcol0 = col0 + lr;      if (bcol0 > Nt - 1) bcol0 = Nt - 1;
  int bcol1 = col0 + 16 + lr; if (bcol1 > Nt - 1) bcol1 = Nt - 1;
  const float* Arow = A  + (long)z * bA + (long)arow * lda;
  const float* Bc0  = Bm + (long)z * bB + bcol0;
  const float* Bc1  = Bm + (long)z * bB + bcol1;
  float*       Ob   = O  + (long)z * bO;

  v8f c0, c1;
#pragma unroll
  for (int r = 0; r < 8; ++r) { c0[r] = 0.0f; c1[r] = 0.0f; }

  if (two) {
    for (int k0 = 0; k0 < K; k0 += 32) {
      if (k0 + 32 < K)
        __builtin_prefetch(Bc0 + (long)(k0 + 32 + kb) * ldb, 0, 1);
      v16bf a = load_frag_rowmajor(Arow, k0 + ka);
      v16bf b0, b1;
#pragma unroll
      for (int e = 0; e < 16; ++e) {
        long kr = (long)(k0 + kb + e) * ldb;
        b0[e] = f2bf(Bc0[kr]);
        b1[e] = f2bf(Bc1[kr]);
      }
      c0 = __builtin_amdgcn_wmma_f32_16x16x32_bf16(false, a, false, b0,
                                                   (short)0, c0, false, false);
      c1 = __builtin_amdgcn_wmma_f32_16x16x32_bf16(false, a, false, b1,
                                                   (short)0, c1, false, false);
    }
  } else {
    for (int k0 = 0; k0 < K; k0 += 32) {
      if (k0 + 32 < K)
        __builtin_prefetch(Bc0 + (long)(k0 + 32 + kb) * ldb, 0, 1);
      v16bf a = load_frag_rowmajor(Arow, k0 + ka);
      v16bf b0;
#pragma unroll
      for (int e = 0; e < 16; ++e)
        b0[e] = f2bf(Bc0[(long)(k0 + kb + e) * ldb]);
      c0 = __builtin_amdgcn_wmma_f32_16x16x32_bf16(false, a, false, b0,
                                                   (short)0, c0, false, false);
    }
  }

  int rb = (lane & 16) ? 8 : 0;
  int colA = col0 + lr;
  if (colA < Nt) {
#pragma unroll
    for (int r = 0; r < 8; ++r) {
      int row = row0 + rb + r;
      if (row < M) Ob[(long)row * ldo + colA] = c0[r];
    }
  }
  int colB = col0 + 16 + lr;
  if (two && colB < Nt) {
#pragma unroll
    for (int r = 0; r < 8; ++r) {
      int row = row0 + rb + r;
      if (row < M) Ob[(long)row * ldo + colB] = c1[r];
    }
  }
}

// ---------------------------------------------------------------------------
// Rotate/translate q-points and kv-points into global frames.
// q/k point outputs land in K=32-padded rows: qpt/kpt[(i*NH+h)*32 + p*3+x]
// (pads zeroed by k_norms). v-points keep tight layout for the att@vp GEMM.
// slot: 0..7 -> q-points, 8..15 -> k-points, 16..27 -> v-points
// ---------------------------------------------------------------------------
__global__ void k_transform(const float* __restrict__ qpp,
                            const float* __restrict__ kvpp,
                            const float* __restrict__ rot,
                            const float* __restrict__ trans,
                            float* __restrict__ qpt,
                            float* __restrict__ kpt,
                            float* __restrict__ vpt)
{
  int idx = blockIdx.x * blockDim.x + threadIdx.x;
  if (idx >= NRES * NH * 28) return;
  int slot = idx % 28;
  int h = (idx / 28) % NH;
  int i = idx / (28 * NH);
  float in[3];
  if (slot < PQK) {
    int pp = slot;
#pragma unroll
    for (int x = 0; x < 3; ++x) in[x] = qpp[i*192 + (x*NH + h)*PQK + pp];
  } else {
    int pp = slot - PQK;   // 0..19 within (PQK+PV)
#pragma unroll
    for (int x = 0; x < 3; ++x) in[x] = kvpp[i*480 + (x*NH + h)*(PQK+PV) + pp];
  }
  float out[3];
#pragma unroll
  for (int xi = 0; xi < 3; ++xi) {
    float acc = trans[i*3 + xi];
#pragma unroll
    for (int xj = 0; xj < 3; ++xj) acc += rot[i*9 + xi*3 + xj] * in[xj];
    out[xi] = acc;
  }
  if (slot < PQK) {
    int pp = slot;
#pragma unroll
    for (int x = 0; x < 3; ++x) qpt[(i*NH + h)*32 + pp*3 + x] = out[x];
  } else if (slot < 2*PQK) {
    int pp = slot - PQK;
#pragma unroll
    for (int x = 0; x < 3; ++x) kpt[(i*NH + h)*32 + pp*3 + x] = out[x];
  } else {
    int pp = slot - 2*PQK;
#pragma unroll
    for (int x = 0; x < 3; ++x) vpt[((i*NH + h)*PV + pp)*3 + x] = out[x];
  }
}

// Row-wise squared norms + zero the K pads (slots 24..31) for the K=32 GEMM.
__global__ void k_norms(float* __restrict__ qpt, float* __restrict__ kpt,
                        float* __restrict__ qn, float* __restrict__ kn)
{
  int idx = blockIdx.x * blockDim.x + threadIdx.x;
  if (idx >= NRES * NH) return;
  float a = 0.f, b = 0.f;
#pragma unroll
  for (int e = 0; e < PQK*3; ++e) {
    float q = qpt[idx*32 + e]; a += q*q;
    float k = kpt[idx*32 + e]; b += k*k;
  }
#pragma unroll
  for (int e = PQK*3; e < 32; ++e) { qpt[idx*32 + e] = 0.f; kpt[idx*32 + e] = 0.f; }
  qn[idx] = a; kn[idx] = b;
}

// ---------------------------------------------------------------------------
// Fused logit assembly + softmax, in place in att[h,i,:].
// logit = qk(prescaled) + sqrt(1/3)*bias_p + pt_att + INF*(m2-1)
// pt_att = -0.5*hw[h]*(|q|^2 + |k|^2 - 2 q.k)
// ---------------------------------------------------------------------------
__global__ __launch_bounds__(256) void k_softmax(
    float* __restrict__ att, const float* __restrict__ biasp,
    const float* __restrict__ ptd, const float* __restrict__ qn,
    const float* __restrict__ kn, const float* __restrict__ head_w,
    const float* __restrict__ mask)
{
  __shared__ float red[256];
  int r = blockIdx.x;
  int h = r / NRES, i = r % NRES;
  int t = threadIdx.x;
  float hw = logf(1.0f + expf(head_w[h])) * 0.09622504486493763f; // sqrt(1/108)
  float qni = qn[i*NH + h];
  float mi  = mask[i];
  const float c3 = 0.5773502691896258f; // sqrt(1/3)
  float*       row = att + ((long)h*NRES + i) * NRES;
  const float* ptr = ptd + ((long)h*NRES + i) * NRES;
  float vals[3];
  float lmax = -3.0e38f;
#pragma unroll
  for (int q = 0; q < 3; ++q) {
    int j = t + q*256;
    float v = row[j];
    v += c3 * biasp[((long)i*NRES + j)*NH + h];
    float d2 = qni + kn[j*NH + h] - 2.0f * ptr[j];
    v -= 0.5f * hw * d2;
    v += 100000.0f * (mi * mask[j] - 1.0f);
    vals[q] = v;
    lmax = fmaxf(lmax, v);
  }
  red[t] = lmax; __syncthreads();
  for (int s2 = 128; s2 > 0; s2 >>= 1) {
    if (t < s2) red[t] = fmaxf(red[t], red[t + s2]);
    __syncthreads();
  }
  float m = red[0]; __syncthreads();
  float lsum = 0.f;
#pragma unroll
  for (int q = 0; q < 3; ++q) { vals[q] = expf(vals[q] - m); lsum += vals[q]; }
  red[t] = lsum; __syncthreads();
  for (int s2 = 128; s2 > 0; s2 >>= 1) {
    if (t < s2) red[t] += red[t + s2];
    __syncthreads();
  }
  float inv = 1.0f / red[0];
#pragma unroll
  for (int q = 0; q < 3; ++q) row[t + q*256] = vals[q] * inv;
}

// ---------------------------------------------------------------------------
// Back-rotate attended points, compute norms, scatter into concat buffer.
// concat per row i: [o 0..1023 | flat 1024..1311 | norm 1312..1407 | opair 1408..2431]
// o and opair were written directly by the GEMMs.
// ---------------------------------------------------------------------------
__global__ void k_concat(const float* __restrict__ optb,
                         const float* __restrict__ rot,
                         const float* __restrict__ trans,
                         float* __restrict__ concat)
{
  int idx = blockIdx.x * blockDim.x + threadIdx.x;
  if (idx >= NRES * NH * PV) return;
  int pv = idx % PV;
  int h  = (idx / PV) % NH;
  int i  = idx / (PV * NH);
  float d[3];
#pragma unroll
  for (int x = 0; x < 3; ++x)
    d[x] = optb[i*(NH*PV*3) + h*(PV*3) + pv*3 + x] - trans[i*3 + x];
  float ob[3];
#pragma unroll
  for (int xi = 0; xi < 3; ++xi) {
    float acc = 0.f;
#pragma unroll
    for (int xj = 0; xj < 3; ++xj) acc += rot[i*9 + xj*3 + xi] * d[xj]; // R^T
    ob[xi] = acc;
  }
  float nrm = sqrtf(ob[0]*ob[0] + ob[1]*ob[1] + ob[2]*ob[2] + 1e-8f);
  float* c = concat + (long)i * CONCAT_DIM;
#pragma unroll
  for (int x = 0; x < 3; ++x) c[1024 + x*(NH*PV) + h*PV + pv] = ob[x];
  c[1312 + h*PV + pv] = nrm;
}

// LayerNorm of (a+b) over 256 features; optional duplicate store to out2.
__global__ __launch_bounds__(256) void k_ln(
    const float* __restrict__ a, const float* __restrict__ b,
    const float* __restrict__ g, const float* __restrict__ be,
    float* __restrict__ out, float* __restrict__ out2)
{
  __shared__ float red[256];
  int i = blockIdx.x, t = threadIdx.x;
  float x = a[i*CS + t] + b[i*CS + t];
  red[t] = x; __syncthreads();
  for (int s2 = 128; s2 > 0; s2 >>= 1) {
    if (t < s2) red[t] += red[t + s2];
    __syncthreads();
  }
  float mu = red[0] * (1.0f/CS); __syncthreads();
  float d = x - mu;
  red[t] = d * d; __syncthreads();
  for (int s2 = 128; s2 > 0; s2 >>= 1) {
    if (t < s2) red[t] += red[t + s2];
    __syncthreads();
  }
  float rs = rsqrtf(red[0] * (1.0f/CS) + 1e-5f);
  float o = d * rs * g[t] + be[t];
  out[i*CS + t] = o;
  if (out2) out2[i*CS + t] = o;
}

// Quaternion frame update: rot_new = rot @ quat2rot(norm(1,u0,u1,u2)),
// trans_new = trans + rot @ u[3:6]
__global__ void k_finish(const float* __restrict__ upd,
                         const float* __restrict__ mask,
                         const float* __restrict__ rot,
                         const float* __restrict__ trans,
                         float* __restrict__ out_rot,
                         float* __restrict__ out_trans)
{
  int i = blockIdx.x * blockDim.x + threadIdx.x;
  if (i >= NRES) return;
  float mk = mask[i];
  float u[6];
#pragma unroll
  for (int k = 0; k < 6; ++k) u[k] = upd[i*6 + k] * mk;
  float n = rsqrtf(1.0f + u[0]*u[0] + u[1]*u[1] + u[2]*u[2]);
  float w = n, x = u[0]*n, y = u[1]*n, z = u[2]*n;
  float r[9] = {
    1.f-2.f*(y*y+z*z), 2.f*(x*y-w*z),     2.f*(x*z+w*y),
    2.f*(x*y+w*z),     1.f-2.f*(x*x+z*z), 2.f*(y*z-w*x),
    2.f*(x*z-w*y),     2.f*(y*z+w*x),     1.f-2.f*(x*x+y*y)};
  const float* R = rot + i*9;
#pragma unroll
  for (int a = 0; a < 3; ++a)
#pragma unroll
    for (int cix = 0; cix < 3; ++cix) {
      float acc = 0.f;
#pragma unroll
      for (int k = 0; k < 3; ++k) acc += R[a*3 + k] * r[k*3 + cix];
      out_rot[i*9 + a*3 + cix] = acc;
    }
#pragma unroll
  for (int a = 0; a < 3; ++a) {
    float acc = trans[i*3 + a];
#pragma unroll
    for (int j = 0; j < 3; ++j) acc += R[a*3 + j] * u[3 + j];
    out_trans[i*3 + a] = acc;
  }
}

// ---------------------------------------------------------------------------
// Host side
// ---------------------------------------------------------------------------
static inline long tile_count(int M, int Nt) {
  long ntl = (Nt + 15) / 16;
  long ntp = (ntl + 1) / 2;
  return (long)((M + 15) / 16) * ntp;
}

static inline void run_wt(hipStream_t st, const float* A, long lda, long bA,
                          const float* W, long ldw, long bW, const float* bias,
                          float* O, long ldo, long bO, int M, int Nt, int K,
                          float scale, int relu, int batches) {
  int blocks = (int)((tile_count(M, Nt) + 7) / 8);
  gemm_wt<<<dim3(blocks, 1, batches), dim3(256), 0, st>>>(
      A, lda, bA, W, ldw, bW, bias, O, ldo, bO, M, Nt, K, scale, relu);
}

static inline void run_bn(hipStream_t st, const float* A, long lda, long bA,
                          const float* Bm, long ldb, long bB,
                          float* O, long ldo, long bO, int M, int Nt, int K,
                          int batches) {
  int blocks = (int)((tile_count(M, Nt) + 7) / 8);
  gemm_bn<<<dim3(blocks, 1, batches), dim3(256), 0, st>>>(
      A, lda, bA, Bm, ldb, bB, O, ldo, bO, M, Nt, K);
}

extern "C" void kernel_launch(void* const* d_in, const int* in_sizes, int n_in,
                              void* d_out, int out_size, void* d_ws, size_t ws_size,
                              hipStream_t stream) {
  (void)in_sizes; (void)n_in; (void)out_size; (void)ws_size;
  const float* s     = (const float*)d_in[0];
  const float* p     = (const float*)d_in[1];
  const float* rot   = (const float*)d_in[2];
  const float* trans = (const float*)d_in[3];
  const float* mask  = (const float*)d_in[4];
  const float* Wq  = (const float*)d_in[5];   const float* bq  = (const float*)d_in[6];
  const float* Wkv = (const float*)d_in[7];   const float* bkv = (const float*)d_in[8];
  const float* Wqp = (const float*)d_in[9];   const float* bqp = (const float*)d_in[10];
  const float* Wkvp= (const float*)d_in[11];  const float* bkvp= (const float*)d_in[12];
  const float* Wpb = (const float*)d_in[13];  const float* bpb = (const float*)d_in[14];
  const float* Wo  = (const float*)d_in[15];  const float* bo  = (const float*)d_in[16];
  const float* Wt1 = (const float*)d_in[17];  const float* bt1 = (const float*)d_in[18];
  const float* Wt2 = (const float*)d_in[19];  const float* bt2 = (const float*)d_in[20];
  const float* Wt3 = (const float*)d_in[21];  const float* bt3 = (const float*)d_in[22];
  const float* Wbb = (const float*)d_in[23];  const float* bbb = (const float*)d_in[24];
  const float* head_w = (const float*)d_in[25];
  const float* g1  = (const float*)d_in[26];  const float* be1 = (const float*)d_in[27];
  const float* g2  = (const float*)d_in[28];  const float* be2 = (const float*)d_in[29];

  float* ws = (float*)d_ws;
  float* qb    = ws;                     // 768*1024
  float* kvb   = qb    + 786432L;        // 768*2048
  float* qpp   = kvb   + 1572864L;       // 768*192
  float* kvpp  = qpp   + 147456L;        // 768*480
  float* qptp  = kvpp  + 368640L;        // 768*8*32 (K=32 padded)
  float* kptp  = qptp  + 196608L;        // 768*8*32 (K=32 padded)
  float* vpt   = kptp  + 196608L;        // 768*8*12*3
  float* qn    = vpt   + 221184L;        // 6144
  float* kn    = qn    + 6144L;          // 6144
  float* biasp = kn    + 6144L;          // 768*768*8
  float* att   = biasp + 4718592L;       // 8*768*768 (logits -> att in place)
  float* ptd   = att   + 4718592L;       // 8*768*768
  float* conc  = ptd   + 4718592L;       // 768*2432
  float* optb  = conc  + 1867776L;       // 768*8*12*3
  float* ipa   = optb  + 221184L;        // 768*256
  float* s1    = ipa   + 196608L;        // 768*256
  float* x1    = s1    + 196608L;        // 768*256
  float* x2    = x1    + 196608L;        // 768*256
  float* t3o   = x2    + 196608L;        // 768*256
  float* s2w   = t3o   + 196608L;        // 768*256
  float* updb  = s2w   + 196608L;        // 768*6

  float* out   = (float*)d_out;
  float* out_s2    = out;
  float* out_rot   = out + 196608L;
  float* out_trans = out + 203520L;

  const float qk_scale = 0.05103103630798288f; // sqrt(1/(3*128))

  // 1) projections from s
  run_wt(stream, s, CS, 0, Wq,   CS, 0, bq,   qb,   NH*CH, 0, NRES, NH*CH,         CS, 1.f, 0, 1);
  run_wt(stream, s, CS, 0, Wkv,  CS, 0, bkv,  kvb,  2048,  0, NRES, 2*NH*CH,       CS, 1.f, 0, 1);
  run_wt(stream, s, CS, 0, Wqp,  CS, 0, bqp,  qpp,  192,   0, NRES, NH*PQK*3,      CS, 1.f, 0, 1);
  run_wt(stream, s, CS, 0, Wkvp, CS, 0, bkvp, kvpp, 480,   0, NRES, NH*(PQK+PV)*3, CS, 1.f, 0, 1);

  // 2) point transforms + norms (also zero-fills K pads of qptp/kptp)
  k_transform<<<dim3((NRES*NH*28 + 255)/256), dim3(256), 0, stream>>>(
      qpp, kvpp, rot, trans, qptp, kptp, vpt);
  k_norms<<<dim3((NRES*NH + 255)/256), dim3(256), 0, stream>>>(qptp, kptp, qn, kn);

  // 3) bias_p = p @ Wpb.T + bpb   (M = N*N = 589824, Nt = 8, K = 128)
  run_wt(stream, p, CZ, 0, Wpb, CZ, 0, bpb, biasp, NH, 0,
         NRES*NRES, NH, CZ, 1.f, 0, 1);

  // 4) per-head q.k logits (prescaled) into att buffer
  run_wt(stream, qb, NH*CH, CH, kvb, 2048, 256, nullptr,
         att, NRES, (long)NRES*NRES, NRES, NRES, CH, qk_scale, 0, NH);

  // 5) per-head point cross dots: ptd[h,i,j] = qpt_i . kpt_j  (K padded to 32)
  run_wt(stream, qptp, NH*32, 32, kptp, NH*32, 32, nullptr,
         ptd, NRES, (long)NRES*NRES, NRES, NRES, 32, 1.f, 0, NH);

  // 6) fused logit assembly + softmax (in place)
  k_softmax<<<dim3(NH*NRES), dim3(256), 0, stream>>>(
      att, biasp, ptd, qn, kn, head_w, mask);

  // 7) o = att @ v  -> concat[:,0:1024]
  run_bn(stream, att, NRES, (long)NRES*NRES, kvb + CH, 2048, 256,
         conc, CONCAT_DIM, CH, NRES, CH, NRES, NH);
  // 8) opt = att @ vp -> optb
  run_bn(stream, att, NRES, (long)NRES*NRES, vpt, NH*PV*3, PV*3,
         optb, NH*PV*3, PV*3, NRES, PV*3, NRES, NH);
  // 9) opair[i,h,:] = att[:,i,:] @ p[i]  -> concat[:,1408:2432] (batched over i)
  run_bn(stream, att, (long)NRES*NRES, NRES, p, CZ, (long)NRES*CZ,
         conc + 1408, CZ, CONCAT_DIM, NH, CZ, NRES, NRES);

  // 10) back-rotate points, norms, scatter into concat[:,1024:1408]
  k_concat<<<dim3((NRES*NH*PV + 255)/256), dim3(256), 0, stream>>>(
      optb, rot, trans, conc);

  // 11) ipa_out = concat @ Wo.T + bo
  run_wt(stream, conc, CONCAT_DIM, 0, Wo, CONCAT_DIM, 0, bo,
         ipa, CS, 0, NRES, CS, CONCAT_DIM, 1.f, 0, 1);

  // 12) s1 = LN(s + ipa)
  k_ln<<<dim3(NRES), dim3(256), 0, stream>>>(s, ipa, g1, be1, s1, nullptr);

  // 13) transition MLP
  run_wt(stream, s1, CS, 0, Wt1, CS, 0, bt1, x1,  CS, 0, NRES, CS, CS, 1.f, 1, 1);
  run_wt(stream, x1, CS, 0, Wt2, CS, 0, bt2, x2,  CS, 0, NRES, CS, CS, 1.f, 1, 1);
  run_wt(stream, x2, CS, 0, Wt3, CS, 0, bt3, t3o, CS, 0, NRES, CS, CS, 1.f, 0, 1);

  // 14) s2 = LN(s1 + t3o) -> ws copy + d_out
  k_ln<<<dim3(NRES), dim3(256), 0, stream>>>(s1, t3o, g2, be2, s2w, out_s2);

  // 15) upd = s2 @ Wbb.T + bbb   (mask applied in k_finish)
  run_wt(stream, s2w, CS, 0, Wbb, CS, 0, bbb, updb, 6, 0, NRES, 6, CS, 1.f, 0, 1);

  // 16) frame update -> rot_new, trans_new
  k_finish<<<dim3((NRES + 255)/256), dim3(256), 0, stream>>>(
      updb, mask, rot, trans, out_rot, out_trans);
}
